// GatedTextPool_58737972740301
// MI455X (gfx1250) — matrix operand order repaired and checked by reading, not compiled, not run on hardware
//
#include <hip/hip_runtime.h>
#include <hip/hip_bf16.h>
#include <math.h>

// ---------------------------------------------------------------------------
// GatedTextPool fused kernel for MI455X (gfx1250, wave32, WMMA)
//   T=262144 tokens, D=512, H=256, N=8192 segments (ids sorted)
// ---------------------------------------------------------------------------

typedef __bf16 bf16_t;
typedef __attribute__((ext_vector_type(16))) __bf16 v16bf;
typedef __attribute__((ext_vector_type(8)))  float  v8f;
typedef int v4i __attribute__((vector_size(16)));   // matches builtin param type

#define T_TOK   262144
#define D_DIM   512
#define H_DIM   256
#define N_SEG   8192
#define TM      128          // tokens per block
#define LDS_ROW 528          // padded row length in bf16 (512 + 16) to spread banks

#if __has_builtin(__builtin_amdgcn_global_load_async_to_lds_b128) && \
    __has_builtin(__builtin_amdgcn_s_wait_asynccnt)
#define HAVE_ASYNC_LDS 1
#else
#define HAVE_ASYNC_LDS 0
#endif

// ---------------------------------------------------------------------------
// Kernel 0: zero the accumulators and build W1^T in bf16 ([H=256][D=512])
// ---------------------------------------------------------------------------
__global__ void __launch_bounds__(256)
prep_kernel(const float* __restrict__ w1,      // [D=512][H=256] row-major
            bf16_t* __restrict__ w1t,          // [256][512] bf16 out
            float* __restrict__ segg,          // [N] gate sums (zeroed)
            float* __restrict__ out)           // [N][D] pooled accumulator (zeroed)
{
    const int i      = blockIdx.x * blockDim.x + threadIdx.x;
    const int stride = gridDim.x * blockDim.x;
    for (int idx = i; idx < N_SEG * D_DIM; idx += stride) out[idx] = 0.0f;
    for (int idx = i; idx < N_SEG; idx += stride) segg[idx] = 0.0f;
    for (int idx = i; idx < H_DIM * D_DIM; idx += stride) {
        const int n = idx >> 9;       // 0..255 (H index)
        const int k = idx & 511;      // 0..511 (D index)
        w1t[idx] = (bf16_t)w1[k * H_DIM + n];
    }
}

// A-fragment loader: 16-bit A 16x32, contiguous k order (k = 16*hi + s).
// lane<16 holds K {0..15}, lane>=16 holds K {16..31} of row M=ln.
__device__ __forceinline__ v16bf load_a_frag(const float* __restrict__ zrow, int kbase)
{
    const float4 f0 = *(const float4*)(zrow + kbase + 0);
    const float4 f1 = *(const float4*)(zrow + kbase + 4);
    const float4 f2 = *(const float4*)(zrow + kbase + 8);
    const float4 f3 = *(const float4*)(zrow + kbase + 12);
    v16bf a;
    a[0]  = (bf16_t)f0.x; a[1]  = (bf16_t)f0.y; a[2]  = (bf16_t)f0.z; a[3]  = (bf16_t)f0.w;
    a[4]  = (bf16_t)f1.x; a[5]  = (bf16_t)f1.y; a[6]  = (bf16_t)f1.z; a[7]  = (bf16_t)f1.w;
    a[8]  = (bf16_t)f2.x; a[9]  = (bf16_t)f2.y; a[10] = (bf16_t)f2.z; a[11] = (bf16_t)f2.w;
    a[12] = (bf16_t)f3.x; a[13] = (bf16_t)f3.y; a[14] = (bf16_t)f3.z; a[15] = (bf16_t)f3.w;
    return a;
}

// ---------------------------------------------------------------------------
// Kernel 1: fused gate MLP (WMMA bf16) + segmented gated accumulation
// ---------------------------------------------------------------------------
__global__ void __launch_bounds__(256, 1)
fused_gate_pool(const float* __restrict__ zt,        // [T][512] f32
                const long long* __restrict__ seg,   // [T] sorted segment ids
                const float* __restrict__ b1,        // [256]
                const float* __restrict__ w2,        // [256]
                const float* __restrict__ b2p,       // [1]
                const bf16_t* __restrict__ w1t,      // [256][512] bf16 (transposed W1)
                float* __restrict__ segg,            // [N]
                float* __restrict__ out)             // [N][512] accumulator
{
    extern __shared__ char smem[];
    bf16_t* sw1 = (bf16_t*)smem;                          // [256][LDS_ROW] bf16
    float*  sg  = (float*)(smem + 256 * LDS_ROW * 2);     // [TM] gate values
    int*    ssg = (int*)(sg + TM);                        // [TM] segment ids

    const int tid = threadIdx.x;
    const int tb  = blockIdx.x * TM;

    // stage segment ids (int64 -> int32)
    if (tid < TM) ssg[tid] = (int)seg[tb + tid];

    // stage W1^T into LDS (256 KB payload), coalesced 16B chunks.
    // CDNA5 path: async global->LDS copies tracked by ASYNCcnt.
#if HAVE_ASYNC_LDS
    for (int idx = tid; idx < (H_DIM * D_DIM) / 8; idx += 256) {
        const int r = idx >> 6;          // row 0..255
        const int c = (idx & 63) << 3;   // col 0..504, step 8 bf16 (16B)
        __builtin_amdgcn_global_load_async_to_lds_b128(
            (__attribute__((address_space(1))) v4i*)(w1t + r * D_DIM + c),
            (__attribute__((address_space(3))) v4i*)(sw1 + r * LDS_ROW + c),
            0, 0);
    }
    __builtin_amdgcn_s_wait_asynccnt(0);
#else
    for (int idx = tid; idx < (H_DIM * D_DIM) / 8; idx += 256) {
        const int r = idx >> 6;
        const int c = (idx & 63) << 3;
        const ulonglong2 v = *(const ulonglong2*)(w1t + r * D_DIM + c);
        *(ulonglong2*)(sw1 + r * LDS_ROW + c) = v;
    }
#endif
    __syncthreads();

    // ----- Phase A: h[16 tokens x 256] per wave via WMMA bf16 -----
    const int wave = tid >> 5;
    const int lane = tid & 31;
    const int ln   = lane & 15;     // A: row M = ln ; B: column N = 16j + ln
    const int hi   = lane >> 4;     // 0: K low half, 1: K high half (contiguous k order used here)
    const float* zrow = zt + (size_t)(tb + wave * 16 + ln) * D_DIM;

    // 8 LDS base pointers, one per pair of N-tiles; second row of the pair and
    // the per-k column offset both fit the 16-bit ds immediate once the k-loop
    // is fully unrolled -> no address VALU between WMMAs, no hazard NOPs.
    const bf16_t* bp[8];
#pragma unroll
    for (int p = 0; p < 8; ++p)
        bp[p] = sw1 + (p * 32 + ln) * LDS_ROW + hi * 16;

    v8f acc[16];
#pragma unroll
    for (int j = 0; j < 16; ++j) acc[j] = (v8f)0.0f;

    v16bf a_cur = load_a_frag(zrow, hi * 16);
#pragma unroll
    for (int kk = 0; kk < 16; ++kk) {           // K = 512 in steps of 32
        v16bf a_nxt;
        if (kk < 15)                             // prefetch next A under the WMMAs
            a_nxt = load_a_frag(zrow, (kk + 1) * 32 + hi * 16);

        // B double-buffer: issue loads for tile j+1 before the WMMA of tile j
        // so the ds latency hides under the matrix op (distinct VGPR buffers).
        v16bf bcur = *(const v16bf*)(bp[0] + kk * 32);
#pragma unroll
        for (int j = 0; j < 16; ++j) {
            v16bf bnxt;
            if (j < 15) {
                const int jn = j + 1;
                bnxt = *(const v16bf*)(bp[jn >> 1] + (jn & 1) * (16 * LDS_ROW) + kk * 32);
            }
            acc[j] = __builtin_amdgcn_wmma_f32_16x16x32_bf16(
                         false, a_cur, false, bcur, (short)0, acc[j], false, false);
            bcur = bnxt;
        }
        a_cur = a_nxt;
    }

    // ----- Epilogue: g = sigmoid(relu(h + b1) . w2 + b2) -----
    // C layout: VGPR r of lane holds h[M = hi*8 + r][N = 16j + ln] for tile j.
    const float b2 = b2p[0];
    float racc[8];
#pragma unroll
    for (int r = 0; r < 8; ++r) racc[r] = 0.0f;
#pragma unroll
    for (int j = 0; j < 16; ++j) {
        const float b1v = b1[j * 16 + ln];
        const float w2v = w2[j * 16 + ln];
#pragma unroll
        for (int r = 0; r < 8; ++r) {
            float h = acc[j][r] + b1v;
            h = h > 0.0f ? h : 0.0f;
            racc[r] += h * w2v;
        }
    }
    // reduce over the 16 lanes of each half-wave (columns)
#pragma unroll
    for (int r = 0; r < 8; ++r) {
#pragma unroll
        for (int m = 1; m < 16; m <<= 1)
            racc[r] += __shfl_xor(racc[r], m, 32);
    }
    if (ln == 0) {
#pragma unroll
        for (int r = 0; r < 8; ++r) {
            const float x = racc[r] + b2;
            sg[wave * 16 + hi * 8 + r] = 1.0f / (1.0f + __expf(-x));
        }
    }
    __syncthreads();

    // ----- Phase B: segmented accumulation (ids sorted => run-length flush) -----
    // per-segment gate sum (128 LDS reads, few atomics)
    if (tid == 0) {
        int cur = ssg[0];
        float s = 0.0f;
        for (int t = 0; t < TM; ++t) {
            const int sid = ssg[t];
            if (sid != cur) { atomicAdd(&segg[cur], s); s = 0.0f; cur = sid; }
            s += sg[t];
        }
        atomicAdd(&segg[cur], s);
    }

    // each thread owns 2 columns; zt tile re-read hits L2/WGP$ (just streamed)
    {
        const int c0 = tid * 2;
        int cur = ssg[0];
        float s0 = 0.0f, s1 = 0.0f;
        for (int t = 0; t < TM; ++t) {
            const int sid = ssg[t];
            if (sid != cur) {
                atomicAdd(&out[(size_t)cur * D_DIM + c0 + 0], s0);
                atomicAdd(&out[(size_t)cur * D_DIM + c0 + 1], s1);
                s0 = 0.0f; s1 = 0.0f; cur = sid;
            }
            const float  gv = sg[t];
            const float2 z  = *(const float2*)(zt + (size_t)(tb + t) * D_DIM + c0);
            s0 += gv * z.x;
            s1 += gv * z.y;
        }
        atomicAdd(&out[(size_t)cur * D_DIM + c0 + 0], s0);
        atomicAdd(&out[(size_t)cur * D_DIM + c0 + 1], s1);
    }
}

// ---------------------------------------------------------------------------
// Kernel 2: out[n] = p / max(||p||, 1e-12),  p = pooled[n] / (seg_g[n] + 1e-6)
// ---------------------------------------------------------------------------
__global__ void __launch_bounds__(256)
finalize_kernel(const float* __restrict__ segg, float* __restrict__ out)
{
    __shared__ float red[8];
    const int n   = blockIdx.x;
    const int tid = threadIdx.x;
    float* row = out + (size_t)n * D_DIM;

    const float scale = 1.0f / (segg[n] + 1e-6f);
    const float p0 = row[tid]       * scale;
    const float p1 = row[tid + 256] * scale;

    float part = p0 * p0 + p1 * p1;
#pragma unroll
    for (int m = 16; m >= 1; m >>= 1) part += __shfl_xor(part, m, 32);
    if ((tid & 31) == 0) red[tid >> 5] = part;
    __syncthreads();
    float tot = 0.0f;
#pragma unroll
    for (int i = 0; i < 8; ++i) tot += red[i];

    const float inv = 1.0f / fmaxf(sqrtf(tot), 1e-12f);
    row[tid]       = p0 * inv;
    row[tid + 256] = p1 * inv;
}

// ---------------------------------------------------------------------------
extern "C" void kernel_launch(void* const* d_in, const int* in_sizes, int n_in,
                              void* d_out, int out_size, void* d_ws, size_t ws_size,
                              hipStream_t stream)
{
    const float*     zt  = (const float*)d_in[0];
    const long long* seg = (const long long*)d_in[1];
    const float*     w1  = (const float*)d_in[2];
    const float*     b1  = (const float*)d_in[3];
    const float*     w2  = (const float*)d_in[4];
    const float*     b2  = (const float*)d_in[5];
    float*           out = (float*)d_out;

    // workspace: [0,256KB) W1^T bf16 ; then seg_g floats
    bf16_t* w1t  = (bf16_t*)d_ws;
    float*  segg = (float*)((char*)d_ws + (size_t)H_DIM * D_DIM * sizeof(bf16_t));

    prep_kernel<<<4096, 256, 0, stream>>>(w1, w1t, segg, out);

    const size_t lds_bytes = 256 * LDS_ROW * 2      // W1^T staged, padded
                           + TM * sizeof(float)     // gate values
                           + TM * sizeof(int);      // segment ids
    fused_gate_pool<<<T_TOK / TM, 256, lds_bytes, stream>>>(
        zt, seg, b1, w2, b2, w1t, segg, out);

    finalize_kernel<<<N_SEG, 256, 0, stream>>>(segg, out);
}
